// Gru_proposal_network_49168785605084
// MI455X (gfx1250) — compile-verified, matrix operand order checked by reading, not run on hardware
//
#include <hip/hip_runtime.h>
#include <hip/hip_bf16.h>

// ---------------------------------------------------------------------------
// GRU proposal network for MI455X (gfx1250), wave32 + WMMA.
// W_ih + r/z-gate W_hh register-resident (304 VGPRs) across all 32 steps;
// n-gate W_hh streams from L2 each step (LICM defeated via opaque zero).
// Hidden state in LDS as f16.
// ---------------------------------------------------------------------------

typedef __attribute__((ext_vector_type(16))) _Float16 v16h;
typedef __attribute__((ext_vector_type(8)))  _Float16 v8h;
typedef __attribute__((ext_vector_type(8)))  float    v8f;

#define SEQ   32
#define ROWS  2048
#define FDIM  16
#define HID   256

// packed weight layout in d_ws (halves):
//   whhP : 48 ntiles x 8 ktiles x 512 halves  = 196608 halves (384 KB)
//   wihP : 48 ntiles x 512 halves             =  24576 halves ( 48 KB)
//   then b_ih (768 f32), b_hh (768 f32)
#define WHH_HALFS (48 * 8 * 512)
#define WIH_HALFS (48 * 512)

__device__ __forceinline__ float fsigmoid(float x) { return 1.0f / (1.0f + __expf(-x)); }
__device__ __forceinline__ float ftanh_fast(float x) {
    float e = __expf(2.0f * x);
    return (e - 1.0f) / (e + 1.0f);
}
__device__ __forceinline__ float fsoftplus(float x) {
    return x > 20.0f ? x : __logf(1.0f + __expf(x));
}

__device__ __forceinline__ v8f wmma16(v16h a, v16h b, v8f c) {
    // v_wmma_f32_16x16x32_f16  (A 16x32 f16, B 32x16 f16, C/D 16x16 f32)
    return __builtin_amdgcn_wmma_f32_16x16x32_f16(
        /*neg_a=*/false, a, /*neg_b=*/false, b,
        /*c_mod=*/(short)0, c, /*reuse_a=*/false, /*reuse_b=*/false);
}

__device__ __forceinline__ v16h loadBtile(const _Float16* p) {
    // 32-byte aligned contiguous per-lane fragment -> 2x global_load_b128
    return *(const v16h*)p;
}

// ---------------------------------------------------------------------------
// Prepack: f32 weights -> f16 WMMA B-fragment layout.
// K-index striping for 16-bit fragments (ISA 7.12.2):
//   lane<16 : halves 0..7 -> K 0..7,   halves 8..15 -> K 16..23
//   lane>=16: halves 0..7 -> K 8..15,  halves 8..15 -> K 24..31
// ---------------------------------------------------------------------------
__global__ void gru_prepack(const float* __restrict__ Wih, const float* __restrict__ Whh,
                            const float* __restrict__ bih, const float* __restrict__ bhh,
                            _Float16* __restrict__ whhP, _Float16* __restrict__ wihP,
                            float* __restrict__ bihW, float* __restrict__ bhhW) {
    int idx = blockIdx.x * 256 + threadIdx.x;
    if (idx < WHH_HALFS) {
        int h    = idx & 15;
        int lane = (idx >> 4) & 31;
        int tile = idx >> 9;          // ntG*8 + kt
        int kt   = tile & 7;
        int ntG  = tile >> 3;
        int n    = lane & 15;
        int hi   = lane >> 4;
        int k    = ((h < 8) ? 0 : 16) + (hi ? 8 : 0) + (h & 7);
        int g    = ntG * 16 + n;      // output feature (0..767)
        whhP[idx] = (_Float16)Whh[g * HID + kt * 32 + k];
    } else if (idx < WHH_HALFS + WIH_HALFS) {
        int r    = idx - WHH_HALFS;
        int h    = r & 15;
        int lane = (r >> 4) & 31;
        int ntG  = r >> 9;
        int n    = lane & 15;
        int hi   = lane >> 4;
        int k    = ((h < 8) ? 0 : 16) + (hi ? 8 : 0) + (h & 7);
        int g    = ntG * 16 + n;
        wihP[r] = (_Float16)Wih[g * 32 + k];
    } else if (idx < WHH_HALFS + WIH_HALFS + 768) {
        int g = idx - WHH_HALFS - WIH_HALFS;
        bihW[g] = bih[g];
        bhhW[g] = bhh[g];
    }
}

// ---------------------------------------------------------------------------
// Main fused GRU kernel. 128 blocks x 16 rows, 8 waves x 32 hidden columns.
// ---------------------------------------------------------------------------
__global__ __launch_bounds__(256, 1) void gru_main(
    const float* __restrict__ data, const float* __restrict__ mask,
    const _Float16* __restrict__ whhP, const _Float16* __restrict__ wihP,
    const float* __restrict__ bihW, const float* __restrict__ bhhW,
    float* __restrict__ out) {

    __shared__ _Float16 hbuf[16 * HID];   // 16 rows x 256 hidden, f16, 8 KB

    const int lane    = threadIdx.x & 31;
    const int wave    = threadIdx.x >> 5;
    const int rowBase = blockIdx.x << 4;   // 16-row tile
    const int jBase   = wave << 5;         // 32 hidden columns per wave
    const int nlo     = lane & 15;
    const int hi      = lane >> 4;

    // zero initial hidden state
    for (int i = threadIdx.x; i < 16 * HID; i += 256) hbuf[i] = (_Float16)0.0f;

    // N-tile indices in the 768-wide gate stack: r(tt0,tt1), z(tt0,tt1), n(tt0,tt1)
    const int jb4 = jBase >> 4;
    int ntArr[6];
    ntArr[0] = jb4;          ntArr[1] = jb4 + 1;
    ntArr[2] = 16 + jb4;     ntArr[3] = 16 + jb4 + 1;
    ntArr[4] = 32 + jb4;     ntArr[5] = 32 + jb4 + 1;

    // per-lane C-matrix bias init (C layout: column N = lane&15, all 8 rows equal)
    float initRZ[4], initHN[2], initIN[2];
#pragma unroll
    for (int idx = 0; idx < 4; ++idx) {
        int g = ntArr[idx] * 16 + nlo;
        initRZ[idx] = bihW[g] + bhhW[g];   // r/z: gi+gh fused, both biases
    }
#pragma unroll
    for (int tt = 0; tt < 2; ++tt) {
        int g = ntArr[4 + tt] * 16 + nlo;
        initHN[tt] = bhhW[g];              // n-gate hidden path keeps b_hh (scaled by r)
        initIN[tt] = bihW[g];              // n-gate input path keeps b_ih
    }

    // 32-bit per-lane B-fragment offsets
    int offHH[6], offIH[6];
#pragma unroll
    for (int idx = 0; idx < 6; ++idx) {
        offHH[idx] = (ntArr[idx] * 8) * 512 + lane * 16;
        offIH[idx] = ntArr[idx] * 512 + lane * 16;
    }

    // ---- register-resident weights, loaded ONCE, reused by all 32 steps ----
    // W_ih: 6 fragments (48 VGPRs); W_hh r/z gates: 32 fragments (256 VGPRs).
    v16h wIH[6];
#pragma unroll
    for (int idx = 0; idx < 6; ++idx)
        wIH[idx] = loadBtile(wihP + offIH[idx]);

    v16h wRZ[4][8];
#pragma unroll
    for (int idx = 0; idx < 4; ++idx)
#pragma unroll
        for (int kt = 0; kt < 8; ++kt)
            wRZ[idx][kt] = loadBtile(whhP + offHH[idx] + kt * 512);

    __syncthreads();

    const size_t MEANS_OFF   = 0;
    const size_t SCALES_OFF  = 4194304;   // 32*2048*16*4
    const size_t CONTEXT_OFF = 8388608;   // 2 * 4194304

#pragma unroll 1
    for (int t = 0; t < SEQ; ++t) {
        // Opaque zero, regenerated every iteration: keeps the n-gate weight
        // loads loop-VARIANT so LICM cannot hoist them out of the t-loop
        // (which previously turned the L2 stream into scratch spill reloads).
        int tz;
        asm volatile("s_mov_b32 %0, 0" : "=s"(tz));

        // ---- accumulators (init from biases) ----
        v8f accRZ0, accRZ1, accRZ2, accRZ3, accHN0, accHN1, accIN0, accIN1;
#pragma unroll
        for (int i = 0; i < 8; ++i) {
            accRZ0[i] = initRZ[0]; accRZ1[i] = initRZ[1];
            accRZ2[i] = initRZ[2]; accRZ3[i] = initRZ[3];
            accHN0[i] = initHN[0]; accHN1[i] = initHN[1];
            accIN0[i] = initIN[0]; accIN1[i] = initIN[1];
        }

        // ---- input projection gi: A = phi tile (16x32 f16), resident W_ih ----
        {
            const int row = rowBase + nlo;
            const float* dptr = data + ((size_t)t * ROWS + row) * FDIM + (hi ? 8 : 0);
            const float* mptr = mask + ((size_t)t * ROWS + row) * FDIM + (hi ? 8 : 0);
            float4 d0 = *(const float4*)(dptr);
            float4 d1 = *(const float4*)(dptr + 4);
            float4 m0 = *(const float4*)(mptr);
            float4 m1 = *(const float4*)(mptr + 4);
            v16h aX;
            aX[0]  = (_Float16)d0.x; aX[1]  = (_Float16)d0.y;
            aX[2]  = (_Float16)d0.z; aX[3]  = (_Float16)d0.w;
            aX[4]  = (_Float16)d1.x; aX[5]  = (_Float16)d1.y;
            aX[6]  = (_Float16)d1.z; aX[7]  = (_Float16)d1.w;
            aX[8]  = (_Float16)m0.x; aX[9]  = (_Float16)m0.y;
            aX[10] = (_Float16)m0.z; aX[11] = (_Float16)m0.w;
            aX[12] = (_Float16)m1.x; aX[13] = (_Float16)m1.y;
            aX[14] = (_Float16)m1.z; aX[15] = (_Float16)m1.w;

            accRZ0 = wmma16(aX, wIH[0], accRZ0);
            accRZ1 = wmma16(aX, wIH[1], accRZ1);
            accRZ2 = wmma16(aX, wIH[2], accRZ2);
            accRZ3 = wmma16(aX, wIH[3], accRZ3);
            accIN0 = wmma16(aX, wIH[4], accIN0);
            accIN1 = wmma16(aX, wIH[5], accIN1);
        }

        // ---- recurrent projection gh: 8 K-tiles; A fragment from f16 LDS ----
#pragma unroll
        for (int kt = 0; kt < 8; ++kt) {
            int base = nlo * HID + kt * 32 + (hi ? 8 : 0);
            union { v8h h2[2]; v16h v; } ua;
            ua.h2[0] = *(const v8h*)&hbuf[base];
            ua.h2[1] = *(const v8h*)&hbuf[base + 16];
            v16h a = ua.v;

            // n-gate weights stream from L2 (tz makes the address loop-variant)
            v16h bn0 = loadBtile(whhP + (offHH[4] + tz) + kt * 512);
            v16h bn1 = loadBtile(whhP + (offHH[5] + tz) + kt * 512);

            accRZ0 = wmma16(a, wRZ[0][kt], accRZ0);
            accRZ1 = wmma16(a, wRZ[1][kt], accRZ1);
            accRZ2 = wmma16(a, wRZ[2][kt], accRZ2);
            accRZ3 = wmma16(a, wRZ[3][kt], accRZ3);
            accHN0 = wmma16(a, bn0, accHN0);
            accHN1 = wmma16(a, bn1, accHN1);
        }

        __syncthreads();   // all LDS reads of h done before anyone writes h_new

        // ---- gates, state update, outputs ----
        // (previous-h read + h_new write touch ONLY this wave's 32 columns,
        //  so they are race-free after the barrier)
#pragma unroll
        for (int tt = 0; tt < 2; ++tt) {
            const int colf = (jBase + tt * 16) >> 4;   // feature group f_ (0..15)
            const int c    = nlo;                      // channel within group (0..15)
            const v8f* aR = (tt == 0) ? &accRZ0 : &accRZ1;
            const v8f* aZ = (tt == 0) ? &accRZ2 : &accRZ3;
            const v8f* aN = (tt == 0) ? &accHN0 : &accHN1;
            const v8f* aI = (tt == 0) ? &accIN0 : &accIN1;
#pragma unroll
            for (int i = 0; i < 8; ++i) {
                const int m = hi * 8 + i;              // row within 16-row tile
                const int hidx = m * HID + jBase + tt * 16 + nlo;

                float r = fsigmoid((*aR)[i]);
                float z = fsigmoid((*aZ)[i]);
                float n = ftanh_fast((*aI)[i] + r * (*aN)[i]);
                float hold = (float)hbuf[hidx];
                float hnew = (1.0f - z) * n + z * hold;

                hbuf[hidx] = (_Float16)hnew;

                float hr = fmaxf(hnew, 0.0f);
                size_t ob = ((size_t)t * ROWS + (rowBase + m)) * FDIM + colf;
                if (c < 8) {
                    out[CONTEXT_OFF + ob * 8 + c] = hr;                           // context
                } else if (c < 12) {
                    out[MEANS_OFF + ob * 4 + (c - 8)] = hr;                       // means
                } else {
                    out[SCALES_OFF + ob * 4 + (c - 12)] = fsoftplus(hr) + 0.001f; // scales
                }
            }
        }

        __syncthreads();   // h_new fully written before next step's reads
    }
}

// ---------------------------------------------------------------------------
extern "C" void kernel_launch(void* const* d_in, const int* in_sizes, int n_in,
                              void* d_out, int out_size, void* d_ws, size_t ws_size,
                              hipStream_t stream) {
    const float* data = (const float*)d_in[0];   // [32,2048,16]
    const float* mask = (const float*)d_in[1];   // [32,2048,16]
    const float* Wih  = (const float*)d_in[2];   // [768,32]
    const float* Whh  = (const float*)d_in[3];   // [768,256]
    const float* bih  = (const float*)d_in[4];   // [768]
    const float* bhh  = (const float*)d_in[5];   // [768]

    _Float16* whhP = (_Float16*)d_ws;
    _Float16* wihP = whhP + WHH_HALFS;
    float*    bihW = (float*)(wihP + WIH_HALFS);
    float*    bhhW = bihW + 768;

    // prepack weights into WMMA fragment layout (221952 work items)
    gru_prepack<<<867, 256, 0, stream>>>(Wih, Whh, bih, bhh, whhP, wihP, bihW, bhhW);

    // fused GRU: 2048 rows / 16-row tiles = 128 blocks, 8 waves each
    gru_main<<<128, 256, 0, stream>>>(data, mask, whhP, wihP, bihW, bhhW, (float*)d_out);
}
